// Elman_33629593927790
// MI455X (gfx1250) — matrix operand order repaired and checked by reading, not compiled, not run on hardware
//
#include <hip/hip_runtime.h>
#include <hip/hip_bf16.h>
#include <math.h>

// Elman RNN for MI455X (gfx1250, wave32, WMMA).
// Phases: [prep weights->bf16 fragments] [xproj GEMM] [serial scan over T]
//         [output GEMM in-place].

typedef __attribute__((ext_vector_type(16))) __bf16 v16bf;
typedef __attribute__((ext_vector_type(8)))  float  v8f;

#define D   300
#define DP  320        // padded D (multiple of 32)
#define TT  512
#define BB  128
#define KT  10         // DP / 32 k-steps per WMMA chain
#define NT  20         // DP / 16 n-tiles (one per wave)
#define FRAG_ELEMS (KT * NT * 32 * 16)   // bf16 elems per pre-swizzled weight
#define XPROJ_ELEMS ((size_t)TT * BB * DP)

// A-operand (16x32 bf16) per-lane fragment position for element (m, kk):
// lanes 0-15 hold kk in {0..7,16..23}, lanes 16-31 hold kk in {8..15,24..31};
// VGPR v holds a (even,odd) kk pair -> index i = 2v+parity within the 16 bf16.
__device__ __forceinline__ int a_frag_pos(int m, int kk) {
  int hi   = (kk >> 3) & 1;            // kk in 8-15 / 24-31 -> upper lane half
  int lane = m + (hi << 4);
  int rem  = kk & 0x17;                // clear bit3: rem in {0..7, 16..23}
  int vg   = (rem < 8) ? (rem >> 1) : (4 + ((rem - 16) >> 1));
  return lane * 16 + ((vg << 1) | (kk & 1));
}
// B-operand (32x16 bf16) fragment position for element (kk, n):
// lane = n + 16*((kk>>3)&1), index i = (kk&7) + 8*(kk>=16).  (Used implicitly in prep.)

// ---------------- Phase 0: weights -> bf16 pre-swizzled B fragments --------
__global__ void prep_weights(const float* __restrict__ W1,
                             const float* __restrict__ W2,
                             __bf16* __restrict__ WxF,
                             __bf16* __restrict__ WhF,
                             __bf16* __restrict__ W2F) {
  int idx = blockIdx.x * blockDim.x + threadIdx.x;
  if (idx >= 3 * FRAG_ELEMS) return;
  int w = idx / FRAG_ELEMS;
  int e = idx % FRAG_ELEMS;            // ((kt*NT + nt)*32 + lane)*16 + i
  int i    = e & 15;
  int lane = (e >> 4) & 31;
  int nt   = (e >> 9) % NT;
  int kt   = (e >> 9) / NT;
  int n = nt * 16 + (lane & 15);       // output column
  int k = kt * 32 + ((i & 7) | ((lane & 16) >> 1) | ((i & 8) << 1)); // kk = (i&7) + 8*(lane>=16) + 16*(i>=8)
  float v = 0.f;
  if (n < D && k < D) {
    if (w == 0)      v = W1[n * 600 + k];         // Wx[c][d]   (c=n, d=k)
    else if (w == 1) v = W1[n * 600 + 300 + k];   // Wh[c][d]
    else             v = W2[n * 300 + k];         // W2[c][d]
  }
  __bf16* dst = (w == 0) ? WxF : (w == 1) ? WhF : W2F;
  dst[e] = (__bf16)v;
}

// ---------------- Phase 1: xprojT[t][b][c] = x @ Wx^T + b1 -----------------
__global__ __launch_bounds__(640)
void xproj_kernel(const float* __restrict__ x,
                  const float* __restrict__ b1,
                  const __bf16* __restrict__ WxF,
                  float* __restrict__ xprojT) {
  __shared__ __align__(16) __bf16 aF[KT * 32 * 16];   // 10 KB
  int tid = threadIdx.x;
  int m0  = blockIdx.x * 16;           // rows ordered m = t*BB + b
  int t0  = m0 / BB;
  int b0  = m0 % BB;
  // stage 16 rows of x (f32 -> bf16) into A-fragment layout, K padded to 320
  for (int s = 0; s < 8; ++s) {
    int e = s * 640 + tid;             // 0..5119 == 16*320
    int r = e / DP;
    int k = e % DP;
    float v = (k < D) ? x[((size_t)(b0 + r) * TT + t0) * D + k] : 0.f;
    aF[(k >> 5) * 512 + a_frag_pos(r, k & 31)] = (__bf16)v;
  }
  __syncthreads();
  int wave = tid >> 5;                 // 0..19 == n-tile
  int lane = tid & 31;
  v8f acc = {};
  for (int kt = 0; kt < KT; ++kt) {
    v16bf a = *(const v16bf*)(aF + kt * 512 + lane * 16);
    v16bf b = *(const v16bf*)(WxF + (((size_t)kt * NT + wave) * 32 + lane) * 16);
    acc = __builtin_amdgcn_wmma_f32_16x16x32_bf16(false, a, false, b,
                                                  (short)0, acc, false, false);
  }
  int n   = lane & 15;
  int col = wave * 16 + n;
  float bv = (col < D) ? b1[col] : 0.f;   // padded cols stay exactly 0
  int mh = (lane >> 4) << 3;              // C layout: m = r + 8*(lane>=16)
  for (int r = 0; r < 8; ++r) {
    xprojT[(size_t)(m0 + r + mh) * DP + col] = acc[r] + bv;
  }
}

// ---------------- Phase 2: serial scan  h_t = xproj_t + h_{t-1} @ Wh^T -----
__global__ __launch_bounds__(640)
void recur_kernel(const float* __restrict__ xprojT,
                  const __bf16* __restrict__ WhF,
                  float* __restrict__ out) {
  extern __shared__ __align__(16) char smem[];
  __bf16* whL = (__bf16*)smem;                           // 204800 B, LDS-pinned
  __bf16* hA  = (__bf16*)(smem + (size_t)FRAG_ELEMS * 2); // 2 x 10240 B
  int tid  = threadIdx.x;
  int lane = tid & 31;
  int wave = tid >> 5;
  int b0   = blockIdx.x * 16;          // this WG owns 16 independent chains
  // copy Wh fragments to LDS once (reused 512 times)
  {
    const uint32_t* src = (const uint32_t*)WhF;
    uint32_t* dst = (uint32_t*)whL;
    for (int i = tid; i < FRAG_ELEMS / 2; i += 640) dst[i] = src[i];
  }
  for (int i = tid; i < 2 * KT * 512; i += 640) hA[i] = (__bf16)0.f; // h0 = 0
  __syncthreads();
  int n    = lane & 15;
  int mh   = (lane >> 4) << 3;
  int col  = wave * 16 + n;            // produced column == K index next step
  int kt2  = col >> 5;
  int kk   = col & 31;
  for (int t = 0; t < TT; ++t) {
    __bf16* hcur = hA + (t & 1) * (KT * 512);
    __bf16* hnxt = hA + ((t + 1) & 1) * (KT * 512);
    // Issue this step's xproj loads *before* the WMMA chain so their VMEM
    // latency overlaps matrix math (they depend only on t), and prefetch the
    // next step's tile into cache (global_prefetch_b8).
    float xp[8];
#pragma unroll
    for (int r = 0; r < 8; ++r) {
      xp[r] = xprojT[((size_t)t * BB + b0 + r + mh) * DP + col];
    }
    if (t + 1 < TT) {
      __builtin_prefetch(&xprojT[((size_t)(t + 1) * BB + b0 + mh) * DP + col], 0, 1);
    }
    v8f acc = {};
    for (int kt = 0; kt < KT; ++kt) {
      v16bf a = *(const v16bf*)(hcur + kt * 512 + lane * 16);
      v16bf b = *(const v16bf*)(whL + (((kt * NT + wave) * 32 + lane) * 16));
      acc = __builtin_amdgcn_wmma_f32_16x16x32_bf16(false, a, false, b,
                                                    (short)0, acc, false, false);
    }
#pragma unroll
    for (int r = 0; r < 8; ++r) {
      int m = r + mh;
      float hv = acc[r] + xp[r];
      if (col < D) {
        out[((size_t)(b0 + m) * TT + t) * D + col] = hv;   // h_t (pre-act)
        if (t == TT - 1)                                    // final hidden
          out[(size_t)BB * TT * D + (size_t)(b0 + m) * D + col] = hv;
      }
      hnxt[kt2 * 512 + a_frag_pos(m, kk)] = (__bf16)hv;     // feed next step
    }
    __syncthreads();   // double-buffered h: one barrier per step suffices
  }
}

// ---------------- Phase 3: out = sigmoid(h) @ W2^T + b2  (in place) --------
__global__ __launch_bounds__(640)
void out_kernel(const __bf16* __restrict__ W2F,
                const float* __restrict__ b2,
                float* __restrict__ out) {
  __shared__ __align__(16) __bf16 aF[KT * 32 * 16];
  int tid = threadIdx.x;
  size_t m0 = (size_t)blockIdx.x * 16;   // rows ordered m = b*TT + t
  for (int s = 0; s < 8; ++s) {
    int e = s * 640 + tid;
    int r = e / DP;
    int k = e % DP;
    float v = 0.f;
    if (k < D) {
      float hv = out[(m0 + r) * D + k];
      v = 1.f / (1.f + __expf(-hv));
    }
    aF[(k >> 5) * 512 + a_frag_pos(r, k & 31)] = (__bf16)v;
  }
  __syncthreads();   // all reads done before any in-place write below
  int wave = tid >> 5, lane = tid & 31;
  v8f acc = {};
  for (int kt = 0; kt < KT; ++kt) {
    v16bf a = *(const v16bf*)(aF + kt * 512 + lane * 16);
    v16bf b = *(const v16bf*)(W2F + (((size_t)kt * NT + wave) * 32 + lane) * 16);
    acc = __builtin_amdgcn_wmma_f32_16x16x32_bf16(false, a, false, b,
                                                  (short)0, acc, false, false);
  }
  int n = lane & 15;
  int col = wave * 16 + n;
  if (col < D) {
    float bv = b2[col];
    int mh = (lane >> 4) << 3;
    for (int r = 0; r < 8; ++r) {
      out[(m0 + r + mh) * D + col] = acc[r] + bv;
    }
  }
}

extern "C" void kernel_launch(void* const* d_in, const int* in_sizes, int n_in,
                              void* d_out, int out_size, void* d_ws, size_t ws_size,
                              hipStream_t stream) {
  const float* x  = (const float*)d_in[0];
  const float* W1 = (const float*)d_in[1];
  const float* b1 = (const float*)d_in[2];
  const float* W2 = (const float*)d_in[3];
  const float* b2 = (const float*)d_in[4];
  float* out = (float*)d_out;

  char* ws = (char*)d_ws;
  float*  xprojT = (float*)ws;                                   // 83.9 MB
  __bf16* WxF = (__bf16*)(ws + XPROJ_ELEMS * sizeof(float));     // 200 KB each
  __bf16* WhF = WxF + FRAG_ELEMS;
  __bf16* W2F = WhF + FRAG_ELEMS;

  prep_weights<<<(3 * FRAG_ELEMS + 255) / 256, 256, 0, stream>>>(W1, W2, WxF, WhF, W2F);
  xproj_kernel<<<(BB * TT) / 16, 640, 0, stream>>>(x, b1, WxF, xprojT);
  recur_kernel<<<BB / 16, 640, (size_t)FRAG_ELEMS * 2 + 2 * KT * 512 * 2, stream>>>(xprojT, WhF, out);
  out_kernel<<<(BB * TT) / 16, 640, 0, stream>>>(W2F, b2, out);
}
